// GroupedQueryAttention_28123445854775
// MI455X (gfx1250) — compile-verified, matrix operand order checked by reading, not run on hardware
//
#include <hip/hip_runtime.h>

// ---------------------------------------------------------------------------
// GQA forward for MI455X (gfx1250, wave32, WMMA).
// Pipeline: [gemm Q][gemm K][gemm V] -> [rmsnorm+rope Q][rmsnorm+rope K]
//           -> [transpose V] -> [flash attention, LDS-free] -> [gemm O]
// All matrix math on v_wmma_f32_16x16x32_bf16 (fp32 accumulate).
// Intermediates (~90MB) live in d_ws and fit in the 192MB L2.
// Attention computes S^T = K x Q^T so the softmax axis is in-thread;
// the 1/sqrt(D) scale and the exp->exp2 conversion factor are pre-folded
// into Q, and causal masking is peeled to the diagonal block only.
// ---------------------------------------------------------------------------

typedef __attribute__((ext_vector_type(16))) __bf16 v16bf;
typedef __attribute__((ext_vector_type(8)))  __bf16 v8bf;
typedef __attribute__((ext_vector_type(8)))  float  v8f;

#define EMBD 2048
#define NHQ  32
#define NGK  8
#define HDIM 64
#define GSZ  4      // NHQ / NGK
#define SEQ  2048
#define NBAT 2
#define MROWS (NBAT*SEQ)   // 4096

__device__ __forceinline__ v8f wmma_bf16(v16bf a, v16bf b, v8f c) {
  // 8 args: (neg_a, A, neg_b, B, c_mod, C, reuse_a, reuse_b)
  return __builtin_amdgcn_wmma_f32_16x16x32_bf16(false, a, false, b, (short)0, c,
                                                 false, false);
}

// hardware exp2: single v_exp_f32
__device__ __forceinline__ float fast_exp2(float x) {
  return __builtin_amdgcn_exp2f(x);
}

// 16-bit A/B fragment (ISA 7.12.2): lane holds row (A) / col (B) = lane&15,
// with K elements for group g = lane>>4 in two contiguous 8-element runs:
// K in [8g, 8g+8) and [16+8g, 16+8g+8). rowp points at K=0 of this row.
__device__ __forceinline__ v16bf load_frag(const __bf16* rowp, int g) {
  v8bf lo = *(const v8bf*)(rowp + 8 * g);
  v8bf hi = *(const v8bf*)(rowp + 16 + 8 * g);
  v16bf f;
#pragma unroll
  for (int i = 0; i < 8; ++i) { f[i] = lo[i]; f[i + 8] = hi[i]; }
  return f;
}

// ---------------------------------------------------------------------------
// Generic GEMM: C[M,N] = A[M,K] * B[K,N].  A is fp32 or bf16, B is fp32,
// both converted to bf16 while staging to LDS.  Block tile 128x64x32,
// 8 waves, each wave owns a 32x32 output tile (2x2 WMMA accumulators).
// ---------------------------------------------------------------------------
#define BM 128
#define BN 64
#define BK 32
#define LDT (BK + 8)   // +16B pad: keeps 16B alignment, staggers banks

template <typename AT, typename OT>
__global__ void __launch_bounds__(256)
gemm_bf16_wmma(const AT* __restrict__ A, const float* __restrict__ Bw,
               OT* __restrict__ C, int M, int N, int K) {
  __shared__ __align__(16) __bf16 As[BM][LDT];
  __shared__ __align__(16) __bf16 Bs[BN][LDT];   // transposed: [n][k]
  const int tid  = threadIdx.x;
  const int lane = tid & 31, wave = tid >> 5;
  const int wm   = wave & 3, wn = wave >> 2;      // 4x2 wave grid
  const int bm   = blockIdx.x * BM, bn = blockIdx.y * BN;
  const int g    = lane >> 4, cn = lane & 15;

  v8f acc[2][2] = {};

  for (int k0 = 0; k0 < K; k0 += BK) {
    {  // stage A tile: 128x32, two threads per row
      int r = tid >> 1, c = (tid & 1) * 16;
      const AT* src = A + (size_t)(bm + r) * K + (k0 + c);
      __builtin_prefetch(src + BK, 0, 1);          // global_prefetch next tile
#pragma unroll
      for (int i = 0; i < 16; ++i) As[r][c + i] = (__bf16)(float)src[i];
    }
    {  // stage B tile transposed: 32(k) x 64(n)
      int kk = tid >> 3, c = (tid & 7) * 8;
      const float* src = Bw + (size_t)(k0 + kk) * N + (bn + c);
      __builtin_prefetch(src + (size_t)BK * N, 0, 1);
#pragma unroll
      for (int i = 0; i < 8; ++i) Bs[c + i][kk] = (__bf16)src[i];
    }
    __syncthreads();

    v16bf af[2], bf[2];
#pragma unroll
    for (int im = 0; im < 2; ++im) af[im] = load_frag(&As[wm * 32 + im * 16 + cn][0], g);
#pragma unroll
    for (int in = 0; in < 2; ++in) bf[in] = load_frag(&Bs[wn * 32 + in * 16 + cn][0], g);
#pragma unroll
    for (int im = 0; im < 2; ++im)
#pragma unroll
      for (int in = 0; in < 2; ++in)
        acc[im][in] = wmma_bf16(af[im], bf[in], acc[im][in]);
    __syncthreads();
  }

  // C/D layout: VGPR j, lane l -> row = j + 8*(l>>4), col = l&15
#pragma unroll
  for (int im = 0; im < 2; ++im)
#pragma unroll
    for (int in = 0; in < 2; ++in) {
      int row0 = bm + wm * 32 + im * 16 + 8 * g;
      int col  = bn + wn * 32 + in * 16 + cn;
#pragma unroll
      for (int j = 0; j < 8; ++j)
        C[(size_t)(row0 + j) * N + col] = (OT)acc[im][in][j];
    }
}

// ---------------------------------------------------------------------------
// Fused RMSNorm + RoPE over D=64, one wave per (token,head).
// Lane owns d and d+32 -> rotate-half needs no cross-lane exchange.
// omul folds the attention score scale into Q (0.125 * log2(e)).
// ---------------------------------------------------------------------------
__global__ void __launch_bounds__(256)
rmsnorm_rope(const float* __restrict__ X, __bf16* __restrict__ Y,
             const float* __restrict__ scale,
             const float* __restrict__ cosb, const float* __restrict__ sinb,
             int nheads, float omul) {
  const int lane = threadIdx.x & 31;
  const int idx  = blockIdx.x * 8 + (threadIdx.x >> 5);
  const int row  = idx / nheads;
  const int h    = idx - row * nheads;
  const int s    = row & (SEQ - 1);

  const float* xp = X + (size_t)row * (nheads * HDIM) + h * HDIM;
  float x1 = xp[lane], x2 = xp[lane + 32];
  float ss = x1 * x1 + x2 * x2;
#pragma unroll
  for (int m = 1; m < 32; m <<= 1) ss += __shfl_xor(ss, m, 32);
  float r  = rsqrtf(ss * (1.0f / 64.0f) + 1e-6f);
  float n1 = x1 * r * scale[lane];
  float n2 = x2 * r * scale[lane + 32];
  float c1 = cosb[s * HDIM + lane],      s1 = sinb[s * HDIM + lane];
  float c2 = cosb[s * HDIM + lane + 32], s2 = sinb[s * HDIM + lane + 32];
  __bf16* yp = Y + (size_t)row * (nheads * HDIM) + h * HDIM;
  yp[lane]      = (__bf16)((n1 * c1 - n2 * s1) * omul);
  yp[lane + 32] = (__bf16)((n2 * c2 + n1 * s2) * omul);
}

// ---------------------------------------------------------------------------
// One-time V transpose: Vb[token][g*64+d] -> Vt[(b*NGK+g)*HDIM + d][s]
// so attention can load V^T A-fragments straight from global/L2.
// ---------------------------------------------------------------------------
__global__ void __launch_bounds__(256)
transpose_v(const __bf16* __restrict__ Vb, __bf16* __restrict__ Vt) {
  __shared__ __bf16 t[HDIM][HDIM + 8];
  const int bg = blockIdx.y;             // b*NGK + g
  const int s0 = blockIdx.x * 64;
  const int tid = threadIdx.x;
  {
    int sl = tid >> 2, d0 = (tid & 3) * 16;
    int b = bg >> 3, g = bg & 7;
    const __bf16* src = Vb + ((size_t)(b * SEQ + s0 + sl)) * (NGK * HDIM) + g * HDIM + d0;
#pragma unroll
    for (int i = 0; i < 16; ++i) t[d0 + i][sl] = src[i];
  }
  __syncthreads();
  {
    int dl = tid >> 2, sS = (tid & 3) * 16;
    __bf16* dst = Vt + ((size_t)bg * HDIM + dl) * SEQ + s0 + sS;
#pragma unroll
    for (int i = 0; i < 16; ++i) dst[i] = t[dl][sS + i];
  }
}

// ---------------------------------------------------------------------------
// One 32-key block of the online-softmax loop.  MASKED is true only for the
// diagonal block (kb == q0); all earlier blocks are fully below the causal
// boundary so the compare/cndmask chains vanish from the steady-state loop.
// Scores arrive pre-scaled by 0.125*log2(e) (folded into Q) -> use exp2.
// ---------------------------------------------------------------------------
template <bool MASKED>
__device__ __forceinline__ void attn_block(
    int kb, int q0, int g, int cn,
    const __bf16* __restrict__ kbase, const __bf16* __restrict__ vbase,
    const v16bf (&qb)[2][2],
    float (&rmax)[2], float (&rsum)[2], v8f (&cacc)[2][4]) {
  // K as A-matrix: rows = keys (two 16-key tiles), K = d
  v16bf ka[2][2];
#pragma unroll
  for (int kt = 0; kt < 2; ++kt) {
    const __bf16* krow = kbase + (size_t)(kb + kt * 16 + cn) * (NGK * HDIM);
    __builtin_prefetch(krow + 32 * NGK * HDIM, 0, 1);   // next key block
#pragma unroll
    for (int kk = 0; kk < 2; ++kk) ka[kt][kk] = load_frag(krow + kk * 32, g);
  }
  // V^T as A-matrix: rows = d (4 tiles), K = 32 keys
  v16bf va[4];
#pragma unroll
  for (int t = 0; t < 4; ++t)
    va[t] = load_frag(vbase + (size_t)(t * 16 + cn) * SEQ + kb, g);

#pragma unroll
  for (int qt = 0; qt < 2; ++qt) {
    v8f s0v = {}, s1v = {};
    s0v = wmma_bf16(ka[0][0], qb[qt][0], s0v);
    s0v = wmma_bf16(ka[0][1], qb[qt][1], s0v);
    s1v = wmma_bf16(ka[1][0], qb[qt][0], s1v);
    s1v = wmma_bf16(ka[1][1], qb[qt][1], s1v);

    float p[16];
    float m = -3.4e38f;
#pragma unroll
    for (int j = 0; j < 8; ++j) {
      float a = s0v[j], c = s1v[j];
      if (MASKED) {
        const int query = q0 + qt * 16 + cn;
        if (kb + j + 8 * g > query)      a = -3.4e38f;
        if (kb + 16 + j + 8 * g > query) c = -3.4e38f;
      }
      p[j] = a; p[8 + j] = c;
      m = fmaxf(m, fmaxf(a, c));
    }
    m = fmaxf(m, __shfl_xor(m, 16, 32));        // combine the two key halves
    float mn   = fmaxf(rmax[qt], m);
    float corr = fast_exp2(rmax[qt] - mn);
    rmax[qt] = mn;

    float psum = 0.f;
    v16bf pb;                                    // P^T B-fragment, in-regs
#pragma unroll
    for (int i = 0; i < 16; ++i) {
      float e = fast_exp2(p[i] - mn);
      psum += e;
      pb[i] = (__bf16)e;
    }
    rsum[qt] = rsum[qt] * corr + psum;

#pragma unroll
    for (int t = 0; t < 4; ++t) {
#pragma unroll
      for (int j = 0; j < 8; ++j) cacc[qt][t][j] *= corr;
      cacc[qt][t] = wmma_bf16(va[t], pb, cacc[qt][t]);
    }
  }
}

// ---------------------------------------------------------------------------
// Flash attention, LDS-free.  One wave per (batch, head, 32-query tile);
// 4 independent waves per block.  exp(S^T) in C-layout is directly the
// B-fragment for ctx^T = V^T x P^T (tile0 8 vals + tile1 8 vals = 16 K elems).
// ---------------------------------------------------------------------------
__global__ void __launch_bounds__(128)
attn_flash(const __bf16* __restrict__ Q, const __bf16* __restrict__ Kc,
           const __bf16* __restrict__ Vt, __bf16* __restrict__ Ctx) {
  const int lane = threadIdx.x & 31;
  const int wave = threadIdx.x >> 5;
  const int g = lane >> 4, cn = lane & 15;
  const int q0 = (blockIdx.x * 4 + wave) * 32;
  const int h = blockIdx.y, b = blockIdx.z;
  const int gk = h / GSZ;

  // Q^T as B-matrix: lane col = query, K = d (two 16-query tiles)
  v16bf qb[2][2];
#pragma unroll
  for (int qt = 0; qt < 2; ++qt) {
    const __bf16* qrow =
        Q + ((size_t)(b * SEQ + q0 + qt * 16 + cn)) * (NHQ * HDIM) + h * HDIM;
#pragma unroll
    for (int kk = 0; kk < 2; ++kk) qb[qt][kk] = load_frag(qrow + kk * 32, g);
  }

  float rmax[2] = {-3.4e38f, -3.4e38f};
  float rsum[2] = {0.f, 0.f};
  v8f cacc[2][4] = {};   // ctx^T: [query-tile][d-tile]; rows=d(j+8g), col=query

  const __bf16* kbase = Kc + ((size_t)(b * SEQ)) * (NGK * HDIM) + gk * HDIM;
  const __bf16* vbase = Vt + ((size_t)(b * NGK + gk) * HDIM) * SEQ;

  const int nfull = q0 / 32;   // blocks fully below the causal boundary
  for (int it = 0; it < nfull; ++it)
    attn_block<false>(it * 32, q0, g, cn, kbase, vbase, qb, rmax, rsum, cacc);
  attn_block<true>(nfull * 32, q0, g, cn, kbase, vbase, qb, rmax, rsum, cacc);

  // finalize: combine key-half partial sums, normalize, store ctx (bf16)
#pragma unroll
  for (int qt = 0; qt < 2; ++qt) {
    float s   = rsum[qt] + __shfl_xor(rsum[qt], 16, 32);
    float inv = 1.0f / s;
    __bf16* crow =
        Ctx + ((size_t)(b * SEQ + q0 + qt * 16 + cn)) * (NHQ * HDIM) + h * HDIM;
#pragma unroll
    for (int t = 0; t < 4; ++t)
#pragma unroll
      for (int j = 0; j < 8; ++j)
        crow[t * 16 + j + 8 * g] = (__bf16)(cacc[qt][t][j] * inv);
  }
}

// ---------------------------------------------------------------------------
extern "C" void kernel_launch(void* const* d_in, const int* in_sizes, int n_in,
                              void* d_out, int out_size, void* d_ws, size_t ws_size,
                              hipStream_t stream) {
  (void)in_sizes; (void)n_in; (void)out_size; (void)ws_size;
  const float* x    = (const float*)d_in[0];
  // d_in[1] = mask (bool) — causality handled analytically
  const float* cosb = (const float*)d_in[2];
  const float* sinb = (const float*)d_in[3];
  const float* Wq   = (const float*)d_in[4];
  const float* Wk   = (const float*)d_in[5];
  const float* Wv   = (const float*)d_in[6];
  const float* Wo   = (const float*)d_in[7];
  const float* qs   = (const float*)d_in[8];
  const float* ks   = (const float*)d_in[9];
  float* out = (float*)d_out;

  // workspace layout (~90 MB)
  char* w = (char*)d_ws;
  float*  Qf  = (float*)w;   w += (size_t)MROWS * 2048 * 4;  // Q pre-norm (fp32)
  float*  Kf  = (float*)w;   w += (size_t)MROWS * 512  * 4;  // K pre-norm (fp32)
  __bf16* Vb  = (__bf16*)w;  w += (size_t)MROWS * 512  * 2;  // V (bf16)
  __bf16* Qb  = (__bf16*)w;  w += (size_t)MROWS * 2048 * 2;  // Q post-rope (bf16)
  __bf16* Kb  = (__bf16*)w;  w += (size_t)MROWS * 512  * 2;  // K post-rope (bf16)
  __bf16* Vt  = (__bf16*)w;  w += (size_t)MROWS * 512  * 2;  // V^T per (b,g)
  __bf16* Cx  = (__bf16*)w;  w += (size_t)MROWS * 2048 * 2;  // attention ctx (bf16)

  dim3 blk(256);
  gemm_bf16_wmma<float, float><<<dim3(MROWS / BM, 2048 / BN), blk, 0, stream>>>(
      x, Wq, Qf, MROWS, 2048, EMBD);
  gemm_bf16_wmma<float, float><<<dim3(MROWS / BM, 512 / BN), blk, 0, stream>>>(
      x, Wk, Kf, MROWS, 512, EMBD);
  gemm_bf16_wmma<float, __bf16><<<dim3(MROWS / BM, 512 / BN), blk, 0, stream>>>(
      x, Wv, Vb, MROWS, 512, EMBD);

  // Q carries the score scale and the exp->exp2 conversion: 0.125 * log2(e)
  rmsnorm_rope<<<dim3(MROWS * NHQ / 8), blk, 0, stream>>>(
      Qf, Qb, qs, cosb, sinb, NHQ, 0.125f * 1.4426950408889634f);
  rmsnorm_rope<<<dim3(MROWS * NGK / 8), blk, 0, stream>>>(
      Kf, Kb, ks, cosb, sinb, NGK, 1.0f);

  transpose_v<<<dim3(SEQ / 64, NBAT * NGK), blk, 0, stream>>>(Vb, Vt);

  attn_flash<<<dim3(SEQ / 128, NHQ, NBAT), dim3(128), 0, stream>>>(Qb, Kb, Vt, Cx);

  gemm_bf16_wmma<__bf16, float><<<dim3(MROWS / BM, 2048 / BN), blk, 0, stream>>>(
      Cx, Wo, out, MROWS, 2048, 2048);
}
